// TransformerLayer_103079215324
// MI455X (gfx1250) — compile-verified, matrix operand order checked by reading, not run on hardware
//
#include <hip/hip_runtime.h>
#include <math.h>
#include <stdint.h>

#define B_ 2
#define S_ 2048
#define D_ 768
#define H_ 12
#define DK_ 64
#define F_ 3072
#define MROWS (B_ * S_)          // 4096
#define QKVW (3 * D_)            // 2304

typedef __attribute__((ext_vector_type(16))) __bf16 v16bf;
typedef __attribute__((ext_vector_type(8)))  __bf16 v8bf;
typedef __attribute__((ext_vector_type(8)))  float  v8f;
typedef unsigned int v4u __attribute__((ext_vector_type(4)));
typedef int          v4i __attribute__((ext_vector_type(4)));
typedef int          v8i __attribute__((ext_vector_type(8)));

// Tensor Data Mover available? (device pass only; host pass parses fallback)
#if defined(__AMDGCN__) && __has_builtin(__builtin_amdgcn_tensor_load_to_lds) && \
    __has_builtin(__builtin_amdgcn_s_wait_tensorcnt)
#define USE_TDM 1
#else
#define USE_TDM 0
#endif

static __device__ __forceinline__ __bf16 f2bf(float x) { return (__bf16)x; }

static __device__ __forceinline__ v8f wmma_bf16(v16bf a, v16bf b, v8f c) {
  return __builtin_amdgcn_wmma_f32_16x16x32_bf16(
      /*neg_a=*/false, a, /*neg_b=*/false, b,
      /*c_mod=*/(short)0, c, /*reuse_a=*/false, /*reuse_b=*/false);
}

// Row-major fragment load (A operand; also B operand from an n-major tile).
// Per ISA 7.12.2 16-bit A 16x32 layout:
//   elems 0..7  -> K = half*8 .. half*8+7
//   elems 8..15 -> K = 16+half*8 .. 16+half*8+7
static __device__ __forceinline__ v16bf frag_row(const __bf16* src, int ld) {
  int lane = threadIdx.x & 31;
  int m = lane & 15;
  int half = lane >> 4;
  const __bf16* p = src + (size_t)m * ld;
  v8bf lo = *(const v8bf*)(p + half * 8);
  v8bf hi = *(const v8bf*)(p + 16 + half * 8);
  v16bf a;
#pragma unroll
  for (int i = 0; i < 8; ++i) { a[i] = lo[i]; a[8 + i] = hi[i]; }
  return a;
}

// Column-style B fragment from a k-major (row-major [k][n]) source.
static __device__ __forceinline__ v16bf frag_col(const __bf16* src, int ld) {
  int lane = threadIdx.x & 31;
  int n = lane & 15;
  int half = lane >> 4;
  v16bf b;
  const __bf16* p0 = src + (size_t)(half * 8) * ld + n;
  const __bf16* p1 = src + (size_t)(16 + half * 8) * ld + n;
#pragma unroll
  for (int e = 0; e < 8; ++e) {
    b[e]     = p0[(size_t)e * ld];
    b[8 + e] = p1[(size_t)e * ld];
  }
  return b;
}

// ---------------- LayerNorm (fp32 in -> bf16 out), one block per row ------
__global__ __launch_bounds__(256) void ln_kernel(
    const float* __restrict__ x, const float* __restrict__ g,
    const float* __restrict__ bb, __bf16* __restrict__ out) {
  __shared__ float red[256];
  int row = blockIdx.x;
  int tid = threadIdx.x;
  const float* xr = x + (size_t)row * D_;
  float v0 = xr[tid], v1 = xr[tid + 256], v2 = xr[tid + 512];
  red[tid] = v0 + v1 + v2;
  __syncthreads();
  for (int st = 128; st > 0; st >>= 1) {
    if (tid < st) red[tid] += red[tid + st];
    __syncthreads();
  }
  float mu = red[0] * (1.0f / D_);
  __syncthreads();
  float d0 = v0 - mu, d1 = v1 - mu, d2 = v2 - mu;
  red[tid] = d0 * d0 + d1 * d1 + d2 * d2;
  __syncthreads();
  for (int st = 128; st > 0; st >>= 1) {
    if (tid < st) red[tid] += red[tid + st];
    __syncthreads();
  }
  float rstd = rsqrtf(red[0] * (1.0f / D_) + 1e-5f);
  __bf16* orow = out + (size_t)row * D_;
  orow[tid]       = f2bf(d0 * rstd * g[tid] + bb[tid]);
  orow[tid + 256] = f2bf(d1 * rstd * g[tid + 256] + bb[tid + 256]);
  orow[tid + 512] = f2bf(d2 * rstd * g[tid + 512] + bb[tid + 512]);
}

// ---------------- Weight packing ------------------------------------------
__global__ __launch_bounds__(256) void pack_qkv_w(
    const float* __restrict__ wq, const float* __restrict__ wk,
    const float* __restrict__ wv, const float* __restrict__ bq,
    const float* __restrict__ bk, const float* __restrict__ bv,
    __bf16* __restrict__ Wqkv, float* __restrict__ biasqkv) {
  int i = blockIdx.x * 256 + threadIdx.x;  // over D_*768
  if (i < D_ * D_) {
    int d = i / D_, c = i % D_;
    int hh = c / DK_, kk = c % DK_;
    size_t src = ((size_t)hh * D_ + d) * DK_ + kk;
    Wqkv[(size_t)d * QKVW + c]          = f2bf(wq[src]);
    Wqkv[(size_t)d * QKVW + D_ + c]     = f2bf(wk[src]);
    Wqkv[(size_t)d * QKVW + 2 * D_ + c] = f2bf(wv[src]);
  }
  if (i < D_) {
    biasqkv[i] = bq[i];
    biasqkv[D_ + i] = bk[i];
    biasqkv[2 * D_ + i] = bv[i];
  }
}

__global__ __launch_bounds__(256) void f2bf_copy(const float* __restrict__ s,
                                                 __bf16* __restrict__ d, int n) {
  int i = blockIdx.x * 256 + threadIdx.x;
  if (i < n) d[i] = f2bf(s[i]);
}

// ---------------- Tiled WMMA GEMM: C = A[MxK] * W[KxN] + epilogue ---------
// EPI 0: +bias, store bf16       (QKV)
// EPI 1: +bias +resid, store f32 (projection / final)
// EPI 2: +bias, GELU, store bf16 (FFN up)
template <int EPI>
__global__ __launch_bounds__(256) void gemm_bf16(
    const __bf16* __restrict__ A, const __bf16* __restrict__ W,
    const float* __restrict__ bias, const float* __restrict__ resid,
    float* __restrict__ outf, __bf16* __restrict__ outb,
    int M, int N, int K) {
  constexpr int BM = 128, BN = 128, BK = 32;
  constexpr int LDA = BK + 16;   // 48 elems -> 96B rows (16B aligned)
  constexpr int LDBT = BK + 16;  // n-major B tile
  __shared__ __bf16 sA[BM * LDA];
  __shared__ __bf16 sB[BN * LDBT];

  int tid = threadIdx.x;
  int wave = tid >> 5;
  int mw = wave & 3;   // 4 m-waves * 32 rows
  int nw = wave >> 2;  // 2 n-waves * 64 cols
  int m0 = blockIdx.y * BM;
  int n0 = blockIdx.x * BN;

  v8f acc[2][4];
#pragma unroll
  for (int i = 0; i < 2; ++i)
#pragma unroll
    for (int j = 0; j < 4; ++j)
#pragma unroll
      for (int e = 0; e < 8; ++e) acc[i][j][e] = 0.0f;

  for (int k0 = 0; k0 < K; k0 += BK) {
#if USE_TDM
    // ---- A tile via Tensor Data Mover: 128x32 bf16, global stride K,
    //      LDS padded to 48-element rows via pad_interval/pad_amount.
    if (wave == 0) {
      unsigned long long ga =
          (unsigned long long)(uintptr_t)(A + (size_t)m0 * K + k0);
      unsigned lds = (unsigned)(uintptr_t)(&sA[0]);
      v4u g0;
      g0[0] = 1u;                                   // count=1, user mode
      g0[1] = lds;                                  // lds_addr
      g0[2] = (unsigned)(ga & 0xFFFFFFFFu);         // global_addr[31:0]
      g0[3] = (unsigned)((ga >> 32) & 0x01FFFFFFu)  // global_addr[56:32]
              | (2u << 30);                         // type=2 ("image")
      v8i g1;
      // data_size=1(2B) | pad_enable | pad_interval=3(16 DW) | pad_amount=7(8 DW)
      g1[0] = (int)((1u << 16) | (1u << 20) | (3u << 22) | (7u << 25));
      g1[1] = (int)(((unsigned)K & 0xFFFFu) << 16);              // tensor_dim0 lo
      g1[2] = (int)((((unsigned)K >> 16) & 0xFFFFu) |
                    (((unsigned)M & 0xFFFFu) << 16));            // dim0 hi|dim1 lo
      g1[3] = (int)((((unsigned)M >> 16) & 0xFFFFu) | (32u << 16)); // dim1 hi|tile0
      g1[4] = 128;                                  // tile_dim1=128 rows
      g1[5] = (int)K;                               // tensor_dim0_stride[31:0]
      g1[6] = 0;
      g1[7] = 0;
      v4i z4 = {0, 0, 0, 0};
#if __clang_major__ >= 23
      v8i z8 = {0, 0, 0, 0, 0, 0, 0, 0};
      __builtin_amdgcn_tensor_load_to_lds(g0, g1, z4, z4, z8, 0);
#else
      __builtin_amdgcn_tensor_load_to_lds(g0, g1, z4, z4, 0);
#endif
      __builtin_amdgcn_s_wait_tensorcnt(0);
    }
#else
    {  // A tile: 128x32, 16 elems/thread, vectorized global->LDS
      int r = tid >> 1, c = (tid & 1) * 16;
      const uint4* g = (const uint4*)(A + (size_t)(m0 + r) * K + k0 + c);
      uint4 u0 = g[0], u1 = g[1];
      uint4* s = (uint4*)(sA + r * LDA + c);
      s[0] = u0; s[1] = u1;
    }
#endif
    {  // B tile: 32x128 loaded row-major, stored n-major (transposed)
      int kk = tid >> 3, c = (tid & 7) * 16;
      const uint4* g = (const uint4*)(W + (size_t)(k0 + kk) * N + n0 + c);
      union { uint4 v[2]; __bf16 e[16]; } u;
      u.v[0] = g[0]; u.v[1] = g[1];
#pragma unroll
      for (int e = 0; e < 16; ++e) sB[(c + e) * LDBT + kk] = u.e[e];
      // prefetch next k-step's B tile into cache hierarchy
      if (k0 + BK < K)
        __builtin_prefetch(W + (size_t)(k0 + BK + kk) * N + n0 + c, 0, 1);
    }
    __syncthreads();

    const __bf16* aBase = sA + (mw * 32) * LDA;
    const __bf16* bBase = sB + (nw * 64) * LDBT;
    v16bf af0 = frag_row(aBase, LDA);
    v16bf af1 = frag_row(aBase + 16 * LDA, LDA);
#pragma unroll
    for (int nn = 0; nn < 4; ++nn) {
      v16bf bf = frag_row(bBase + nn * 16 * LDBT, LDBT);  // lane = n
      acc[0][nn] = wmma_bf16(af0, bf, acc[0][nn]);
      acc[1][nn] = wmma_bf16(af1, bf, acc[1][nn]);
    }
    __syncthreads();
  }

  int lane = tid & 31;
  int col = lane & 15;
  int rb = (lane >> 4) * 8;
#pragma unroll
  for (int mm = 0; mm < 2; ++mm)
#pragma unroll
    for (int nn = 0; nn < 4; ++nn) {
      int gm = m0 + mw * 32 + mm * 16 + rb;
      int gn = n0 + nw * 64 + nn * 16 + col;
      float bsv = bias[gn];
#pragma unroll
      for (int j = 0; j < 8; ++j) {
        float v = acc[mm][nn][j] + bsv;
        size_t idx = (size_t)(gm + j) * N + gn;
        if (EPI == 1) {
          outf[idx] = v + resid[idx];
        } else if (EPI == 2) {
          v = 0.5f * v * (1.0f + erff(v * 0.70710678118654752f));
          outb[idx] = f2bf(v);
        } else {
          outb[idx] = f2bf(v);
        }
      }
    }
}

// ---------------- Flash attention: one wave per (b,h,16 q-rows) -----------
__global__ __launch_bounds__(256) void flash_attn(
    const __bf16* __restrict__ qkv,  // [M][2304]: Q@h*64, K@768+h*64, V@1536+h*64
    __bf16* __restrict__ attout) {   // [M][768]  col h*64
  __shared__ __bf16 pstage[8][16 * 32];
  const int QT = S_ / 16;  // 128
  int wave = threadIdx.x >> 5;
  int gwave = blockIdx.x * 8 + wave;
  int qt = gwave % QT;
  int h = (gwave / QT) % H_;
  int b = gwave / (QT * H_);
  int lane = threadIdx.x & 31;
  int col = lane & 15;
  int rb = (lane >> 4) * 8;

  const size_t ld = QKVW;
  const __bf16* qbase = qkv + ((size_t)b * S_ + qt * 16) * ld + h * DK_;
  const __bf16* kbase = qkv + (size_t)b * S_ * ld + D_ + h * DK_;
  const __bf16* vbase = qkv + (size_t)b * S_ * ld + 2 * D_ + h * DK_;

  v16bf q0 = frag_row(qbase, (int)ld);        // k 0..31
  v16bf q1 = frag_row(qbase + 32, (int)ld);   // k 32..63

  v8f o[4];
#pragma unroll
  for (int nn = 0; nn < 4; ++nn)
#pragma unroll
    for (int e = 0; e < 8; ++e) o[nn][e] = 0.0f;
  float mrun[8], lrun[8];
#pragma unroll
  for (int j = 0; j < 8; ++j) { mrun[j] = -3.0e38f; lrun[j] = 0.0f; }

  const float scale = 0.125f;  // 1/sqrt(64)
  __bf16* ps = pstage[wave];

  for (int s0 = 0; s0 < S_; s0 += 32) {
    v8f sc[2];
#pragma unroll
    for (int sub = 0; sub < 2; ++sub) {
      const __bf16* kb = kbase + (size_t)(s0 + sub * 16) * ld;
      v16bf kf0 = frag_row(kb, (int)ld);       // B = K^T, lane = key index
      v16bf kf1 = frag_row(kb + 32, (int)ld);
      v8f c;
#pragma unroll
      for (int e = 0; e < 8; ++e) c[e] = 0.0f;
      c = wmma_bf16(q0, kf0, c);
      c = wmma_bf16(q1, kf1, c);
      sc[sub] = c;
    }
    // online softmax per row (rows rb..rb+7 live in this half-wave)
    float pA[8], pB[8];
#pragma unroll
    for (int j = 0; j < 8; ++j) {
      float a = sc[0][j] * scale;
      float bvv = sc[1][j] * scale;
      float mx = fmaxf(a, bvv);
#pragma unroll
      for (int d = 1; d < 16; d <<= 1) mx = fmaxf(mx, __shfl_xor(mx, d, 16));
      float mnew = fmaxf(mrun[j], mx);
      float alpha = __expf(mrun[j] - mnew);
      pA[j] = __expf(a - mnew);
      pB[j] = __expf(bvv - mnew);
      float sm = pA[j] + pB[j];
#pragma unroll
      for (int d = 1; d < 16; d <<= 1) sm += __shfl_xor(sm, d, 16);
      lrun[j] = lrun[j] * alpha + sm;
      mrun[j] = mnew;
#pragma unroll
      for (int nn = 0; nn < 4; ++nn) o[nn][j] *= alpha;
    }
    // stage P (C layout -> LDS), reload as A fragment
#pragma unroll
    for (int j = 0; j < 8; ++j) {
      ps[(rb + j) * 32 + col] = f2bf(pA[j]);
      ps[(rb + j) * 32 + 16 + col] = f2bf(pB[j]);
    }
    __syncthreads();
    v16bf pf = frag_row(ps, 32);
#pragma unroll
    for (int nn = 0; nn < 4; ++nn) {
      v16bf vf = frag_col(vbase + (size_t)s0 * ld + nn * 16, (int)ld);
      o[nn] = wmma_bf16(pf, vf, o[nn]);
    }
    __syncthreads();
  }

  __bf16* ob = attout + ((size_t)b * S_ + qt * 16) * D_ + h * DK_;
#pragma unroll
  for (int nn = 0; nn < 4; ++nn)
#pragma unroll
    for (int j = 0; j < 8; ++j) {
      float v = o[nn][j] / lrun[j];
      ob[(size_t)(rb + j) * D_ + nn * 16 + col] = f2bf(v);
    }
}

// ---------------- launcher ------------------------------------------------
extern "C" void kernel_launch(void* const* d_in, const int* in_sizes, int n_in,
                              void* d_out, int out_size, void* d_ws,
                              size_t ws_size, hipStream_t stream) {
  const float* x    = (const float*)d_in[0];
  const float* wq_w = (const float*)d_in[1];
  const float* wq_b = (const float*)d_in[2];
  const float* wk_w = (const float*)d_in[3];
  const float* wk_b = (const float*)d_in[4];
  const float* wv_w = (const float*)d_in[5];
  const float* wv_b = (const float*)d_in[6];
  const float* wo_w = (const float*)d_in[7];
  const float* wo_b = (const float*)d_in[8];
  const float* ln_g = (const float*)d_in[9];
  const float* ln_b = (const float*)d_in[10];
  const float* w1   = (const float*)d_in[11];
  const float* b1   = (const float*)d_in[12];
  const float* w2   = (const float*)d_in[13];
  const float* b2   = (const float*)d_in[14];
  float* out = (float*)d_out;

  char* ws = (char*)d_ws;
  size_t off = 0;
  auto alloc = [&](size_t bytes) -> char* {
    char* p = ws + off;
    off = (off + bytes + 255) & ~(size_t)255;
    return p;
  };
  __bf16* h_bf    = (__bf16*)alloc((size_t)MROWS * D_ * 2);
  __bf16* Wqkv    = (__bf16*)alloc((size_t)D_ * QKVW * 2);
  float*  biasqkv = (float*)alloc((size_t)QKVW * 4);
  __bf16* qkv     = (__bf16*)alloc((size_t)MROWS * QKVW * 2);
  __bf16* att     = (__bf16*)alloc((size_t)MROWS * D_ * 2);
  __bf16* Wo      = (__bf16*)alloc((size_t)D_ * D_ * 2);
  float*  ybuf    = (float*)alloc((size_t)MROWS * D_ * 4);
  __bf16* z_bf    = (__bf16*)alloc((size_t)MROWS * D_ * 2);
  __bf16* W1b     = (__bf16*)alloc((size_t)D_ * F_ * 2);
  __bf16* t_bf    = (__bf16*)alloc((size_t)MROWS * F_ * 2);
  __bf16* W2b     = (__bf16*)alloc((size_t)F_ * D_ * 2);

  // weight packing
  pack_qkv_w<<<(D_ * D_ + 255) / 256, 256, 0, stream>>>(
      wq_w, wk_w, wv_w, wq_b, wk_b, wv_b, Wqkv, biasqkv);
  f2bf_copy<<<(D_ * D_ + 255) / 256, 256, 0, stream>>>(wo_w, Wo, D_ * D_);
  f2bf_copy<<<(D_ * F_ + 255) / 256, 256, 0, stream>>>(w1, W1b, D_ * F_);
  f2bf_copy<<<(F_ * D_ + 255) / 256, 256, 0, stream>>>(w2, W2b, F_ * D_);

  // LN1
  ln_kernel<<<MROWS, 256, 0, stream>>>(x, ln_g, ln_b, h_bf);
  // fused QKV GEMM: [4096 x 2304] = h * Wqkv
  gemm_bf16<0><<<dim3(QKVW / 128, MROWS / 128), 256, 0, stream>>>(
      h_bf, Wqkv, biasqkv, nullptr, nullptr, qkv, MROWS, QKVW, D_);
  // attention
  flash_attn<<<(B_ * H_ * (S_ / 16)) / 8, 256, 0, stream>>>(qkv, att);
  // output projection + residual(x) -> ybuf (f32)
  gemm_bf16<1><<<dim3(D_ / 128, MROWS / 128), 256, 0, stream>>>(
      att, Wo, wo_b, x, ybuf, nullptr, MROWS, D_, D_);
  // LN2
  ln_kernel<<<MROWS, 256, 0, stream>>>(ybuf, ln_g, ln_b, z_bf);
  // FFN up + GELU (bf16)
  gemm_bf16<2><<<dim3(F_ / 128, MROWS / 128), 256, 0, stream>>>(
      z_bf, W1b, b1, nullptr, nullptr, t_bf, MROWS, F_, D_);
  // FFN down + residual(ybuf) -> d_out (f32)
  gemm_bf16<1><<<dim3(D_ / 128, MROWS / 128), 256, 0, stream>>>(
      t_bf, W2b, b2, ybuf, out, nullptr, MROWS, D_, F_);
}
